// _LorentzMultiheadAttention_53687091200452
// MI455X (gfx1250) — matrix operand order repaired and checked by reading, not compiled, not run on hardware
//
#include <hip/hip_runtime.h>
#include <hip/hip_bf16.h>
#include <cstdint>

// ---------------------------------------------------------------------------
// Lorentz multi-head attention for MI455X (gfx1250), wave32 + WMMA bf16 + TDM.
//   1) fp32 -> bf16 convert of src and Wq/Wk/Wv
//   2) QKV projection GEMMs via v_wmma_f32_16x16x32_bf16 (16x64 strip / wave:
//      A-fragment reused across 4 column tiles for 4x less X traffic)
//   3) Lorentz time-coord finalize (pad head dim 65 -> 96, q time negated;
//      V stored transposed [d][s] for contiguous B-fragment loads)
//   4) Flash attention: K/V 32-key chunks staged to LDS once per block by the
//      Tensor Data Mover (tensor_load_to_lds + s_wait_tensorcnt), consumed by
//      4 waves; scores = 3-chunk WMMA chain over K=96; online softmax with
//      half-wave shuffles; probs transposed via per-wave LDS tile; ave via
//      5 N-tile WMMAs; per-head Lorentz renorm epilogue
//   5) head mean + final Lorentz renorm -> out [4,1024,65] fp32
// ---------------------------------------------------------------------------

#define B_  4
#define S_  1024
#define E_  1024
#define H_  16
#define D_  64
#define DP  96   // padded K-dim for qf/k (65 -> 96 = 3 x 32)
#define DV  80   // padded rows for transposed V (65 -> 80 = 5 x 16)
#define BH_ 64   // B_*H_

typedef __attribute__((ext_vector_type(16))) __bf16        v16bf;
typedef __attribute__((ext_vector_type(8)))  float         v8f;
typedef __attribute__((ext_vector_type(4)))  float         f32x4;
typedef __attribute__((ext_vector_type(4)))  unsigned int  u32x4;
typedef __attribute__((ext_vector_type(8)))  int           i32x8;
typedef __attribute__((ext_vector_type(4)))  int           i32x4;

__device__ __forceinline__ unsigned short f2bf(float f) {
  unsigned int u = __float_as_uint(f);
  unsigned int r = u + 0x7FFFu + ((u >> 16) & 1u);   // round-to-nearest-even
  return (unsigned short)(r >> 16);
}
__device__ __forceinline__ float bf2f(unsigned short u) {
  return __uint_as_float(((unsigned int)u) << 16);
}

struct BF16Frag {
  union { v16bf v; f32x4 f4[2]; unsigned short u[16]; };
};

// A-matrix 16x32 bf16 fragment (ISA 7.12.2): lanes 0-15 row M=lane hold
// K = {0..7, 16..23}; lanes 16-31 hold K = {8..15, 24..31}. Two b128 loads.
__device__ __forceinline__ BF16Frag load_frag_A(const unsigned short* p,
                                                int rowStride, int lane) {
  int row = lane & 15, half = lane >> 4;
  const unsigned short* q = p + (size_t)row * rowStride + half * 8;
  BF16Frag f;
  f.f4[0] = *(const f32x4*)(q);
  f.f4[1] = *(const f32x4*)(q + 16);
  return f;
}

// B-matrix 32x16 bf16 fragment from a row-major [N][K] source:
// lane n holds col N=n, K = {half*8 + 0..7, 16 + half*8 + 0..7} (matches A).
__device__ __forceinline__ BF16Frag load_frag_B(const unsigned short* p,
                                                int rowStride, int lane) {
  int n = lane & 15, half = lane >> 4;
  const unsigned short* q = p + (size_t)n * rowStride + half * 8;
  BF16Frag f;
  f.f4[0] = *(const f32x4*)(q);
  f.f4[1] = *(const f32x4*)(q + 16);
  return f;
}

#define WMMA_BF16(a, b, c) \
  __builtin_amdgcn_wmma_f32_16x16x32_bf16(false, (a), false, (b), (short)0, (c), false, false)

// ---------------------------------------------------------------------------
// Tensor Data Mover: 2D tile load Global -> LDS. Descriptor per ISA 8.3/8.4:
// group0 = {count=1, lds_addr, global_addr[56:0], type=2}; group1 packs
// data_size=4B, tensor_dim0/1, tile_dim0/1, tensor_dim0_stride. Groups 2/3
// unused (2D). TENSORcnt-tracked; EXEC ignored -> issue from one wave only.
// This toolchain's builtin takes 6 args (g0, g1, g2, g3, g4, cpol).
__device__ __forceinline__ void tdm_load_2d(unsigned ldsAddr, const void* gptr,
                                            unsigned widthDw, unsigned rows,
                                            unsigned strideDw) {
  unsigned long long ga = (unsigned long long)(uintptr_t)gptr;
  u32x4 g0;
  g0[0] = 1u;                                            // count=1, user D#
  g0[1] = ldsAddr;                                       // lds_addr (bytes)
  g0[2] = (unsigned)(ga & 0xFFFFFFFFu);                  // global_addr lo
  g0[3] = (unsigned)((ga >> 32) & 0x01FFFFFFu) | (2u << 30);  // addr hi | type=2
  i32x8 g1;
  g1[0] = (int)(2u << 16);                               // data_size=2 (4B)
  g1[1] = (int)((widthDw & 0xFFFFu) << 16);              // tensor_dim0[15:0]
  g1[2] = (int)(((widthDw >> 16) & 0xFFFFu) | ((rows & 0xFFFFu) << 16));
  g1[3] = (int)(((rows >> 16) & 0xFFFFu) | ((widthDw & 0xFFFFu) << 16)); // tile_dim0
  g1[4] = (int)(rows & 0xFFFFu);                         // tile_dim1 (tile_dim2=0)
  g1[5] = (int)strideDw;                                 // tensor_dim0_stride lo
  g1[6] = 0;
  g1[7] = 0;
  i32x4 z4 = {0, 0, 0, 0};
  i32x8 z8 = {0, 0, 0, 0, 0, 0, 0, 0};
  __builtin_amdgcn_tensor_load_to_lds(g0, g1, z4, z4, z8, 0);
}

__device__ __forceinline__ unsigned lds_off(const void* p) {
  return (unsigned)(uintptr_t)p;   // low 32 bits of generic addr = LDS offset
}

// ---------------------------------------------------------------------------
__global__ void k_convert(const float* __restrict__ in,
                          unsigned short* __restrict__ out, int n) {
  int i = blockIdx.x * blockDim.x + threadIdx.x;
  int stride = gridDim.x * blockDim.x;
  for (; i < n; i += stride) out[i] = f2bf(in[i]);
}

// ---------------------------------------------------------------------------
// QKV projection: out[row, m] = sum_e X[row,e] * W[m,e].
// One wave computes a 16x64 strip: A-fragment reused across 4 B tiles.
// mode<2 : scatter into padded [bh][s][96] at offset 1+d (q / k)
// mode==2: scatter into transposed [bh][1+d][S] (v)
__global__ __launch_bounds__(32) void k_project(const unsigned short* __restrict__ X,
                                                const unsigned short* __restrict__ W,
                                                unsigned short* __restrict__ dst,
                                                int mode) {
  int lane = threadIdx.x;
  int rt = blockIdx.x;               // row tile: 0..255 (4096 rows)
  int cs = blockIdx.y;               // col strip: 0..15 (4 tiles of 16 each)
  const unsigned short* Xt = X + (size_t)(rt * 16) * E_;
  v8f c[4];
  for (int nt = 0; nt < 4; ++nt) c[nt] = v8f{};
  for (int kb = 0; kb < E_; kb += 32) {
    BF16Frag a = load_frag_A(Xt + kb, E_, lane);
    for (int nt = 0; nt < 4; ++nt) {
      const unsigned short* Wt = W + (size_t)((cs * 4 + nt) * 16) * E_;
      BF16Frag b = load_frag_B(Wt + kb, E_, lane);
      c[nt] = WMMA_BF16(a.v, b.v, c[nt]);
    }
  }
  int half = lane >> 4, nl = lane & 15;
  for (int nt = 0; nt < 4; ++nt) {
    for (int r = 0; r < 8; ++r) {
      int row = rt * 16 + r + half * 8;        // b*S + s
      int col = (cs * 4 + nt) * 16 + nl;       // h*64 + d
      int bb = row >> 10, s = row & 1023;
      int h = col >> 6,  d = col & 63;
      int bh = bb * H_ + h;
      unsigned short val = f2bf(c[nt][r]);
      if (mode < 2) dst[((size_t)bh * S_ + s) * DP + 1 + d] = val;
      else          dst[((size_t)bh * DV + 1 + d) * S_ + s] = val;
    }
  }
}

// ---------------------------------------------------------------------------
// Finalize q/k: time coord t = sqrt(||x||^2 + 1) at index 0 (sign = -1 for q
// so the Lorentz dot becomes a plain dot product), zero the pad [65..95].
__global__ void k_fin_qk(unsigned short* __restrict__ buf, float sign) {
  int idx = blockIdx.x * blockDim.x + threadIdx.x;    // 65536 = 64*1024
  int bh = idx >> 10, s = idx & 1023;
  unsigned short* p = buf + ((size_t)bh * S_ + s) * DP;
  float ss = 0.f;
  for (int i = 1; i <= D_; ++i) { float x = bf2f(p[i]); ss += x * x; }
  float t = sqrtf(ss + 1.0f);
  p[0] = f2bf(sign * t);
  for (int i = D_ + 1; i < DP; ++i) p[i] = 0;
}

// Finalize transposed V: row 0 = time coord, rows 65..79 zeroed.
__global__ void k_fin_v(unsigned short* __restrict__ vt) {
  int idx = blockIdx.x * blockDim.x + threadIdx.x;    // 65536
  int bh = idx >> 10, s = idx & 1023;
  unsigned short* p = vt + (size_t)bh * DV * S_;
  float ss = 0.f;
  for (int d = 1; d <= D_; ++d) { float x = bf2f(p[(size_t)d * S_ + s]); ss += x * x; }
  float t = sqrtf(ss + 1.0f);
  p[s] = f2bf(t);
  for (int d = D_ + 1; d < DV; ++d) p[(size_t)d * S_ + s] = 0;
}

// ---------------------------------------------------------------------------
// Flash attention. 4 waves / block, one wave per 16-row query tile; K/V chunk
// staged into LDS once per block by the TDM, consumed by all 4 waves.
__global__ __launch_bounds__(128) void k_attn(const unsigned short* __restrict__ qf,
                                              const unsigned short* __restrict__ kk,
                                              const unsigned short* __restrict__ vt,
                                              const float* __restrict__ scale,
                                              const float* __restrict__ bias,
                                              float* __restrict__ headOut) {
  __shared__ unsigned short sK[32 * DP];              // 32 keys x 96 dims (6 KB)
  __shared__ unsigned short sV[DV * 32];              // 80 dims x 32 keys (5 KB)
  __shared__ unsigned short pT[4][16 * 32];           // per-wave prob tile (4 KB)
  int lane = threadIdx.x & 31;
  int wave = threadIdx.x >> 5;
  int bh = blockIdx.x;                                // 0..63
  int q0 = (blockIdx.y * 4 + wave) * 16;              // query row base
  int h = bh & (H_ - 1);
  float invScale = 1.0f / scale[h];
  float bi = bias[h];
  const unsigned short* qB = qf + (size_t)bh * S_ * DP;
  const unsigned short* kB = kk + (size_t)bh * S_ * DP;
  const unsigned short* vB = vt + (size_t)bh * DV * S_;
  int half = lane >> 4, nl = lane & 15;
  unsigned short* lds = pT[wave];

  // Query A-fragments for the 3 K-chunks of the padded Lorentz dim (96).
  BF16Frag aq[3];
  for (int c = 0; c < 3; ++c)
    aq[c] = load_frag_A(qB + (size_t)q0 * DP + 32 * c, DP, lane);

  v8f acc[5];
  for (int nt = 0; nt < 5; ++nt) acc[nt] = v8f{};
  float m[8], l[8];
  for (int r = 0; r < 8; ++r) { m[r] = -1e30f; l[r] = 0.f; }

  for (int k0 = 0; k0 < S_; k0 += 32) {
    // ---- stage K (32 rows x 48 dw) and V (80 rows x 16 dw) tiles via TDM ----
    if (wave == 0) {
      tdm_load_2d(lds_off(sK), kB + (size_t)k0 * DP, DP / 2, 32, DP / 2);
      tdm_load_2d(lds_off(sV), vB + k0, 32 / 2, DV, S_ / 2);
      __builtin_amdgcn_s_wait_tensorcnt(0);
    }
    __syncthreads();

    // scores for 32 keys: two 16x16 C tiles, each a 3-chunk WMMA chain
    v8f s0 = {}, s1 = {};
    for (int c = 0; c < 3; ++c) {
      BF16Frag b0 = load_frag_B(sK + 32 * c, DP, lane);
      BF16Frag b1 = load_frag_B(sK + 16 * DP + 32 * c, DP, lane);
      s0 = WMMA_BF16(aq[c].v, b0.v, s0);
      s1 = WMMA_BF16(aq[c].v, b1.v, s1);
    }
    // online softmax; row r lives in vgpr r (+8 for upper lane half)
    for (int r = 0; r < 8; ++r) {
      float lg0 = fmaf(2.0f, s0[r], 2.0f) * invScale + bi;  // (2C + 2s)/scale + b
      float lg1 = fmaf(2.0f, s1[r], 2.0f) * invScale + bi;
      float rmx = fmaxf(lg0, lg1);
      for (int off = 1; off < 16; off <<= 1) rmx = fmaxf(rmx, __shfl_xor(rmx, off, 16));
      float mn = fmaxf(m[r], rmx);
      float f  = __expf(m[r] - mn);
      float p0 = __expf(lg0 - mn);
      float p1 = __expf(lg1 - mn);
      float rsum = p0 + p1;
      for (int off = 1; off < 16; off <<= 1) rsum += __shfl_xor(rsum, off, 16);
      l[r] = l[r] * f + rsum;
      m[r] = mn;
      for (int nt = 0; nt < 5; ++nt) acc[nt][r] *= f;       // rescale accumulator
      int row = r + half * 8;                               // transpose via LDS
      lds[row * 32 + nl]      = f2bf(p0);
      lds[row * 32 + 16 + nl] = f2bf(p1);
    }
    // ave += P(16x32) @ V(32x80): A from per-wave LDS tile, B from staged sV
    BF16Frag ap = load_frag_A(lds, 32, lane);
    for (int nt = 0; nt < 5; ++nt) {
      BF16Frag bv = load_frag_B(sV + (nt * 16) * 32, 32, lane);
      acc[nt] = WMMA_BF16(ap.v, bv.v, acc[nt]);
    }
    __syncthreads();   // all waves done with sK/sV before next TDM stage
  }

  // normalize by softmax denominator
  for (int r = 0; r < 8; ++r) {
    float invl = 1.0f / l[r];
    for (int nt = 0; nt < 5; ++nt) acc[nt][r] *= invl;
  }

  // per-head Lorentz renorm: li = -a0^2 + sum a_i^2 over the 65 valid cols
  float* ho = headOut + (size_t)bh * S_ * 65;
  for (int r = 0; r < 8; ++r) {
    float li = 0.f;
    for (int nt = 0; nt < 5; ++nt) {
      int col = nt * 16 + nl;
      if (col < 65) { float x = acc[nt][r]; li += (col == 0) ? -x * x : x * x; }
    }
    for (int off = 1; off < 16; off <<= 1) li += __shfl_xor(li, off, 16);
    float rn = 1.0f / sqrtf(fmaxf(fabsf(li), 1e-6f));
    int s = q0 + r + half * 8;
    for (int nt = 0; nt < 5; ++nt) {
      int col = nt * 16 + nl;
      if (col < 65) ho[(size_t)s * 65 + col] = acc[nt][r] * rn;
    }
  }
}

// ---------------------------------------------------------------------------
// Head mean + final Lorentz renorm. One wave per (b,s); lane covers cols
// {lane, lane+32, lane+64} < 65.
__global__ __launch_bounds__(256) void k_out(const float* __restrict__ ho,
                                             float* __restrict__ out) {
  int lane = threadIdx.x & 31;
  int wave = threadIdx.x >> 5;
  int idx = blockIdx.x * 8 + wave;          // 0..4095
  int b = idx >> 10, s = idx & 1023;
  float a[3];
  float li = 0.f;
  for (int j = 0; j < 3; ++j) {
    int c = lane + 32 * j;
    a[j] = 0.f;
    if (c < 65) {
      for (int h = 0; h < H_; ++h)
        a[j] += ho[((size_t)(b * H_ + h) * S_ + s) * 65 + c];
      a[j] *= (1.0f / H_);
      li += (c == 0) ? -a[j] * a[j] : a[j] * a[j];
    }
  }
  for (int off = 1; off < 32; off <<= 1) li += __shfl_xor(li, off, 32);
  float rn = 1.0f / sqrtf(fmaxf(fabsf(li), 1e-6f));
  for (int j = 0; j < 3; ++j) {
    int c = lane + 32 * j;
    if (c < 65) out[((size_t)(b * S_ + s)) * 65 + c] = a[j] * rn;
  }
}

// ---------------------------------------------------------------------------
extern "C" void kernel_launch(void* const* d_in, const int* in_sizes, int n_in,
                              void* d_out, int out_size, void* d_ws, size_t ws_size,
                              hipStream_t stream) {
  (void)in_sizes; (void)n_in; (void)out_size; (void)ws_size;
  const float* src   = (const float*)d_in[1];   // source_input (q also uses it)
  const float* Wq    = (const float*)d_in[2];
  const float* Wk    = (const float*)d_in[4];
  const float* Wv    = (const float*)d_in[6];
  const float* scale = (const float*)d_in[8];
  const float* bias  = (const float*)d_in[9];

  char* ws = (char*)d_ws;
  size_t oSrc = 0;
  size_t oW   = oSrc + (size_t)B_ * S_ * E_ * 2;            // src bf16: 8 MB
  size_t oQf  = oW   + 3ull * E_ * E_ * 2;                  // weights bf16: 6 MB
  size_t oKk  = oQf  + (size_t)BH_ * S_ * DP * 2;           // qf: 12 MB
  size_t oVt  = oKk  + (size_t)BH_ * S_ * DP * 2;           // k : 12 MB
  size_t oHo  = oVt  + (size_t)BH_ * DV * S_ * 2;           // vT: 10 MB
                                                            // headOut: 17 MB

  unsigned short* srcB = (unsigned short*)(ws + oSrc);
  unsigned short* w0   = (unsigned short*)(ws + oW);
  unsigned short* w1   = w0 + (size_t)E_ * E_;
  unsigned short* w2   = w1 + (size_t)E_ * E_;
  unsigned short* qfB  = (unsigned short*)(ws + oQf);
  unsigned short* kkB  = (unsigned short*)(ws + oKk);
  unsigned short* vtB  = (unsigned short*)(ws + oVt);
  float*          hoB  = (float*)(ws + oHo);

  k_convert<<<2048, 256, 0, stream>>>(src, srcB, B_ * S_ * E_);
  k_convert<<<1024, 256, 0, stream>>>(Wq, w0, E_ * E_);
  k_convert<<<1024, 256, 0, stream>>>(Wk, w1, E_ * E_);
  k_convert<<<1024, 256, 0, stream>>>(Wv, w2, E_ * E_);

  dim3 gp(256, 16);
  k_project<<<gp, 32, 0, stream>>>(srcB, w0, qfB, 0);
  k_project<<<gp, 32, 0, stream>>>(srcB, w1, kkB, 1);
  k_project<<<gp, 32, 0, stream>>>(srcB, w2, vtB, 2);

  k_fin_qk<<<256, 256, 0, stream>>>(qfB, -1.0f);
  k_fin_qk<<<256, 256, 0, stream>>>(kkB, +1.0f);
  k_fin_v <<<256, 256, 0, stream>>>(vtB);

  dim3 ga(BH_, 16);
  k_attn<<<ga, 128, 0, stream>>>(qfB, kkB, vtB, scale, bias, hoB);

  k_out<<<512, 256, 0, stream>>>(hoB, (float*)d_out);
}
